// VectorQuantizer_22600117912202
// MI455X (gfx1250) — compile-verified, matrix operand order checked by reading, not run on hardware
//
#include <hip/hip_runtime.h>

// ---------------------------------------------------------------------------
// VQ-VAE vector quantizer for MI455X (gfx1250, wave32, WMMA)
//   z_e:      [32, 64, 64, 64] f32   (B, C, H, W)
//   codebook: [1024, 64] f32
// outputs (flat f32): z_q [B,C,H,W] | loss | perplexity | idx [B,H,W]
// ---------------------------------------------------------------------------

typedef __bf16          v16bf __attribute__((ext_vector_type(16)));
typedef float           v8f   __attribute__((ext_vector_type(8)));
typedef unsigned int    v8u   __attribute__((ext_vector_type(8)));
typedef unsigned short  v16us __attribute__((ext_vector_type(16)));

#define NBATCH   32
#define CDIM     64
#define PIX      4096        // H*W
#define NPIX     131072      // NBATCH*PIX
#define NE       1024
#define NTILE_N  64          // NE/16
#define NTILE_M  8192        // NPIX/16

#define ZQ_OFF   0
#define LOSS_OFF 8388608
#define PERP_OFF 8388609
#define IDX_OFF  8388610

// ws layout (u32 units)
#define WS_BHI    0          // 32768 u32 : codebook hi, B-operand swizzled
#define WS_BLO    32768      // 32768 u32 : codebook lo residual
#define WS_EN2    65536      // 1024 f32  : ||e||^2
#define WS_CNT    66560      // 1024 u32  : histogram
#define WS_LOSS   67584      // 1 f32     : SSE accumulator

#define NTHREADS  512        // 16 waves -> 4 waves/SIMD for latency hiding
#define NWAVES    16
#define TILES_PER_WAVE 2     // 256 blocks * 16 waves * 2 = 8192 tiles

__device__ __forceinline__ unsigned short f32_to_bf16_rn(float f) {
  unsigned u = __float_as_uint(f);
  unsigned r = u + 0x7FFFu + ((u >> 16) & 1u);
  return (unsigned short)(r >> 16);
}

// ---------------------------------------------------------------------------
// Prep 1: split codebook into bf16 hi + bf16 lo-residual, pre-swizzled into
// the CDNA5 16-bit B-operand layout (ISA 7.12.2): for code-tile t, K-chunk kc,
// lane l, vgpr v the dword holds bf16 pair (K, K+1) with
//   K = kc*32 + (v<4?0:16) + (l>=16?8:0) + 2*(v&3),  column N = t*16 + (l&15)
// stored lane-major:  word index = ((t*2+kc)*32 + l)*8 + v
// ---------------------------------------------------------------------------
__global__ void vq_prep_b(const float* __restrict__ cb,
                          unsigned* __restrict__ bhi,
                          unsigned* __restrict__ blo) {
  int id = blockIdx.x * 256 + threadIdx.x;   // 0..32767
  int v  = id & 7;
  int l  = (id >> 3) & 31;
  int kc = (id >> 8) & 1;
  int t  = id >> 9;
  int n      = t * 16 + (l & 15);
  int hiHalf = l >> 4;
  int klo    = kc * 32 + (v < 4 ? 0 : 16) + hiHalf * 8 + 2 * (v & 3);

  float f0 = cb[n * CDIM + klo];
  float f1 = cb[n * CDIM + klo + 1];
  unsigned u0 = __float_as_uint(f0) & 0xFFFF0000u;
  unsigned u1 = __float_as_uint(f1) & 0xFFFF0000u;
  unsigned short h0 = (unsigned short)(u0 >> 16);
  unsigned short h1 = (unsigned short)(u1 >> 16);
  unsigned short r0 = f32_to_bf16_rn(f0 - __uint_as_float(u0));
  unsigned short r1 = f32_to_bf16_rn(f1 - __uint_as_float(u1));
  bhi[id] = (unsigned)h0 | ((unsigned)h1 << 16);
  blo[id] = (unsigned)r0 | ((unsigned)r1 << 16);
}

// Prep 2: ||e||^2 per code, zero histogram + loss accumulator (re-done every
// launch so graph replays stay deterministic).
__global__ void vq_prep_misc(const float* __restrict__ cb,
                             float* __restrict__ en2,
                             unsigned* __restrict__ counts,
                             float* __restrict__ lossacc) {
  int c = blockIdx.x * 256 + threadIdx.x;
  if (c < NE) {
    float s = 0.f;
    #pragma unroll 8
    for (int k = 0; k < CDIM; ++k) { float e = cb[c * CDIM + k]; s += e * e; }
    en2[c] = s;
    counts[c] = 0u;
  }
  if (c == 0) *lossacc = 0.f;
}

// ---------------------------------------------------------------------------
// Main kernel: 512 threads (16 waves, 4/SIMD). Stages the full swizzled bf16
// codebook (hi+lo, 256KB) + en2 table into LDS (CDNA5: 320KB/WGP), then each
// wave processes 2 tiles of 16 pixels against all 1024 codes via split-bf16
// WMMA: two independent 3-deep WMMA chains per code tile; 4 waves/SIMD cover
// the WMMA RAW/WAR hazard slots.
// ---------------------------------------------------------------------------
__launch_bounds__(NTHREADS, 1)
__global__ void vq_main(const float* __restrict__ z,
                        const float* __restrict__ cb,
                        const unsigned* __restrict__ gbhi,
                        const unsigned* __restrict__ gblo,
                        const float* __restrict__ gen2,
                        unsigned* __restrict__ counts,
                        float* __restrict__ lossacc,
                        float* __restrict__ out) {
  extern __shared__ unsigned char smem_raw[];
  unsigned* sbhi = (unsigned*)smem_raw;          // 32768 u32
  unsigned* sblo = sbhi + 32768;                 // 32768 u32
  float*    sen2 = (float*)(sblo + 32768);       // 1024 f32
  volatile int* sidx = (int*)(sen2 + NE);        // 16 waves * 16 winners

  const int tid  = threadIdx.x;
  const int lane = tid & 31;
  const int wave = tid >> 5;

  // ---- stage B operands + en2 into LDS -------------------------------
  {
    const v8u* shv = (const v8u*)gbhi;
    const v8u* slv = (const v8u*)gblo;
    v8u* dh = (v8u*)sbhi;
    v8u* dl = (v8u*)sblo;
    for (int i = tid; i < 4096; i += NTHREADS) { dh[i] = shv[i]; dl[i] = slv[i]; }
    for (int i = tid; i < NE; i += NTHREADS)   sen2[i] = gen2[i];
  }
  __syncthreads();

  const int mrow   = lane & 15;   // M row (and score column N-within-tile)
  const int hiHalf = lane >> 4;   // wave32 half: ISA A/C layout split

  const v8u* lbh = (const v8u*)sbhi;
  const v8u* lbl = (const v8u*)sblo;

  for (int it = 0; it < TILES_PER_WAVE; ++it) {
    const int T  = blockIdx.x * (NWAVES * TILES_PER_WAVE)
                 + wave * TILES_PER_WAVE + it;           // pixel tile 0..8191
    const int n0 = T * 16;                               // flat pixel base
    const int b  = T >> 8;                               // 256 tiles / image
    const int p0 = (T & 255) << 4;
    const float* zb = z + (size_t)b * (CDIM * PIX) + p0 + mrow;

    // ---- load A tile (fp32 kept for loss), split into bf16 hi/lo ----
    float zf[32];
    v16bf ahi[2], alo[2];
    #pragma unroll
    for (int kc = 0; kc < 2; ++kc) {
      v16us hb, lb;
      #pragma unroll
      for (int s = 0; s < 16; ++s) {
        const int K = kc * 32 + hiHalf * 8 + (s < 8 ? s : s + 8); // A layout
        const float f = zb[(size_t)K * PIX];
        zf[kc * 16 + s] = f;
        const unsigned u = __float_as_uint(f) & 0xFFFF0000u;
        hb[s] = (unsigned short)(u >> 16);
        lb[s] = f32_to_bf16_rn(f - __uint_as_float(u));
      }
      ahi[kc] = __builtin_bit_cast(v16bf, hb);
      alo[kc] = __builtin_bit_cast(v16bf, lb);
    }

    float bestv[8];
    int   besti[8];
    #pragma unroll
    for (int r = 0; r < 8; ++r) { bestv[r] = 3.0e38f; besti[r] = 0; }

    // ---- sweep all 64 code tiles: 2 independent 3-WMMA chains --------
    for (int t = 0; t < NTILE_N; ++t) {
      const v16bf bh0 = __builtin_bit_cast(v16bf, lbh[(t * 2 + 0) * 32 + lane]);
      const v16bf bl0 = __builtin_bit_cast(v16bf, lbl[(t * 2 + 0) * 32 + lane]);
      const v16bf bh1 = __builtin_bit_cast(v16bf, lbh[(t * 2 + 1) * 32 + lane]);
      const v16bf bl1 = __builtin_bit_cast(v16bf, lbl[(t * 2 + 1) * 32 + lane]);
      v8f accA = {};
      v8f accB = {};
      // chain A (K 0..31) and chain B (K 32..63), interleaved for ILP
      accA = __builtin_amdgcn_wmma_f32_16x16x32_bf16(false, ahi[0], false, bh0,
                                                     (short)0, accA, false, false);
      accB = __builtin_amdgcn_wmma_f32_16x16x32_bf16(false, ahi[1], false, bh1,
                                                     (short)0, accB, false, false);
      accA = __builtin_amdgcn_wmma_f32_16x16x32_bf16(false, alo[0], false, bh0,
                                                     (short)0, accA, false, false);
      accB = __builtin_amdgcn_wmma_f32_16x16x32_bf16(false, alo[1], false, bh1,
                                                     (short)0, accB, false, false);
      accA = __builtin_amdgcn_wmma_f32_16x16x32_bf16(false, ahi[0], false, bl0,
                                                     (short)0, accA, false, false);
      accB = __builtin_amdgcn_wmma_f32_16x16x32_bf16(false, ahi[1], false, bl1,
                                                     (short)0, accB, false, false);

      const int   nidx = t * 16 + mrow;       // this lane's N column
      const float en2v = sen2[nidx];
      #pragma unroll
      for (int r = 0; r < 8; ++r) {           // score = ||e||^2 - 2 z.e
        const float score = __builtin_fmaf(-2.0f, accA[r] + accB[r], en2v);
        if (score < bestv[r]) { bestv[r] = score; besti[r] = nidx; }
      }
    }

    // ---- argmin across the 16 lanes of each half (C-layout M split) --
    #pragma unroll
    for (int r = 0; r < 8; ++r) {
      float bv = bestv[r];
      int   bi = besti[r];
      #pragma unroll
      for (int off = 8; off > 0; off >>= 1) {
        const float ov = __shfl_xor(bv, off, 16);
        const int   oi = __shfl_xor(bi, off, 16);
        if (ov < bv || (ov == bv && oi < bi)) { bv = ov; bi = oi; }  // first-index
      }
      bestv[r] = bv; besti[r] = bi;
    }
    if (mrow == 0) {                          // lane 0 -> rows 0..7, lane 16 -> 8..15
      #pragma unroll
      for (int r = 0; r < 8; ++r) sidx[wave * 16 + hiHalf * 8 + r] = besti[r];
    }
    __builtin_amdgcn_wave_barrier();
    const int myidx = sidx[wave * 16 + mrow];

    // ---- fused: z_q scatter, loss SSE, idx output, histogram ---------
    const float* crow  = cb + (size_t)myidx * CDIM;
    float* zqout = out + ZQ_OFF + (size_t)b * (CDIM * PIX) + p0 + mrow;
    float lsum = 0.f;
    #pragma unroll
    for (int kc = 0; kc < 2; ++kc) {
      #pragma unroll
      for (int s = 0; s < 16; ++s) {
        const int K = kc * 32 + hiHalf * 8 + (s < 8 ? s : s + 8);
        const float e = crow[K];
        const float d = e - zf[kc * 16 + s];
        lsum += d * d;
        zqout[(size_t)K * PIX] = e;           // z + sg(z_q - z) == z_q
      }
    }
    #pragma unroll
    for (int off = 16; off > 0; off >>= 1) lsum += __shfl_xor(lsum, off, 32);
    if (lane == 0) atomicAdd(lossacc, lsum);
    if (lane < 16) {
      const int iv = sidx[wave * 16 + lane];
      out[IDX_OFF + n0 + lane] = (float)iv;
      atomicAdd(&counts[iv], 1u);
    }
    __builtin_amdgcn_wave_barrier();
  }
}

// ---------------------------------------------------------------------------
// Finalize: perplexity from histogram + scaled loss.
// loss = mean((zq-z)^2) + 0.5*mean((zq-z)^2) = 1.5 * SSE / (NPIX*CDIM)
// ---------------------------------------------------------------------------
__global__ void vq_finalize(const unsigned* __restrict__ counts,
                            const float* __restrict__ lossacc,
                            float* __restrict__ out) {
  __shared__ float red[32];
  const int tid = threadIdx.x;                 // 1024 threads
  const float em = (float)counts[tid] * (1.0f / (float)NPIX);
  float term = em * __logf(em + 1e-10f);
  #pragma unroll
  for (int off = 16; off > 0; off >>= 1) term += __shfl_xor(term, off, 32);
  if ((tid & 31) == 0) red[tid >> 5] = term;
  __syncthreads();
  if (tid < 32) {
    float v = red[tid];
    #pragma unroll
    for (int off = 16; off > 0; off >>= 1) v += __shfl_xor(v, off, 32);
    if (tid == 0) {
      out[PERP_OFF] = __expf(-v);
      out[LOSS_OFF] = (*lossacc) * (1.5f / (float)((size_t)NPIX * CDIM));
    }
  }
}

// ---------------------------------------------------------------------------
extern "C" void kernel_launch(void* const* d_in, const int* in_sizes, int n_in,
                              void* d_out, int out_size, void* d_ws, size_t ws_size,
                              hipStream_t stream) {
  const float* z_e = (const float*)d_in[0];
  const float* cb  = (const float*)d_in[1];
  unsigned* ws = (unsigned*)d_ws;
  unsigned* bhi    = ws + WS_BHI;
  unsigned* blo    = ws + WS_BLO;
  float*    en2    = (float*)(ws + WS_EN2);
  unsigned* counts = ws + WS_CNT;
  float*    lossac = (float*)(ws + WS_LOSS);
  float*    out    = (float*)d_out;

  vq_prep_b   <<<128, 256, 0, stream>>>(cb, bhi, blo);
  vq_prep_misc<<<4,   256, 0, stream>>>(cb, en2, counts, lossac);

  const size_t smem = (size_t)32768 * 4 * 2 + (size_t)NE * 4
                    + (size_t)NWAVES * 16 * 4;           // 267,264 B
  vq_main     <<<256, NTHREADS, smem, stream>>>(z_e, cb, bhi, blo, en2,
                                                counts, lossac, out);
  vq_finalize <<<1, 1024, 0, stream>>>(counts, lossac, out);
}
